// GCN_34823594836228
// MI455X (gfx1250) — matrix operand order repaired and checked by reading, not compiled, not run on hardware
//
#include <hip/hip_runtime.h>

#define NN   65536      // nodes
#define NE   1048576    // edges
#define NB   64         // graphs
#define H    128
#define NHEAD 4
#define EPS  1e-5f

typedef __bf16 bf16_t;
typedef __attribute__((ext_vector_type(16))) bf16_t v16bf;
typedef __attribute__((ext_vector_type(8)))  bf16_t v8bf;
typedef __attribute__((ext_vector_type(8)))  float  v8f;

#define CDIV(a,b) (((a)+(b)-1)/(b))

// ---------- helpers ----------
__device__ inline bf16_t f2bf(float f) { return (bf16_t)f; }   // hw v_cvt on gfx1250
__device__ inline unsigned short bf2u(bf16_t b) { return __builtin_bit_cast(unsigned short, b); }
// monotonic float<->uint encoding for atomicMax on floats
__device__ inline unsigned fenc(float f) {
  unsigned u = __builtin_bit_cast(unsigned, f);
  return (u >> 31) ? ~u : (u | 0x80000000u);
}
__device__ inline float fdec(unsigned u) {
  unsigned b = (u & 0x80000000u) ? (u & 0x7FFFFFFFu) : ~u;
  return __builtin_bit_cast(float, b);
}

// ---------- simple memory kernels ----------
__global__ void k_zero(float* p, int n) {
  int g = blockIdx.x * blockDim.x + threadIdx.x;
  if (g < n) p[g] = 0.0f;
}
__global__ void k_cvt_bf16(const float* __restrict__ s, unsigned short* __restrict__ d, int n) {
  int g = blockIdx.x * blockDim.x + threadIdx.x;
  if (g < n) d[g] = bf2u(f2bf(s[g]));
}
__global__ void k_count(const int* __restrict__ gid, float* cnt) {
  int g = blockIdx.x * blockDim.x + threadIdx.x;
  if (g < NN) atomicAdd(cnt + gid[g], 1.0f);
}
__global__ void k_pool(const float* __restrict__ h, const int* __restrict__ gid, float* outp) {
  int g = blockIdx.x * blockDim.x + threadIdx.x;
  if (g >= NN * H) return;
  int n = g >> 7, c = g & 127;
  atomicAdd(outp + gid[n] * H + c, h[g]);
}
__global__ void k_pooldiv(float* outp, const float* cnt) {
  int g = blockIdx.x * blockDim.x + threadIdx.x;
  if (g < NB * H) outp[g] /= fmaxf(cnt[g >> 7], 1.0f);
}
__global__ void k_degree(const int* __restrict__ src, const int* __restrict__ dst,
                         float* doutd, float* dind) {
  int g = blockIdx.x * blockDim.x + threadIdx.x;
  if (g >= NE) return;
  atomicAdd(doutd + src[g], 1.0f);
  atomicAdd(dind  + dst[g], 1.0f);
}
__global__ void k_rsqrt(float* p, int n) {
  int g = blockIdx.x * blockDim.x + threadIdx.x;
  if (g < n) p[g] = rsqrtf(fmaxf(p[g], 1.0f));
}

// ---------- BatchNorm ----------
__global__ void k_bnstats(const float* __restrict__ x, float* stats) {
  int g = blockIdx.x * blockDim.x + threadIdx.x;   // (NN/64)*128 threads
  int c = g & 127;
  int r0 = (g >> 7) * 64;
  float s = 0.f, s2 = 0.f;
  for (int i = 0; i < 64; ++i) {
    float v = x[(size_t)(r0 + i) * H + c];
    s += v; s2 += v * v;
  }
  atomicAdd(stats + c, s);
  atomicAdd(stats + 128 + c, s2);
}
__global__ void k_bnfinal(const float* stats, const float* g_, const float* b_, float* scsh) {
  int c = threadIdx.x;
  float mean = stats[c] * (1.0f / NN);
  float var  = stats[128 + c] * (1.0f / NN) - mean * mean;
  float sc = g_[c] * rsqrtf(var + EPS);
  scsh[c] = sc;
  scsh[128 + c] = b_[c] - mean * sc;
}
// xnorm * rsqrt(deg_out[node]) -> A
__global__ void k_bnapply(const float* __restrict__ x, const float* __restrict__ scsh,
                          const float* __restrict__ dout_r, float* __restrict__ A) {
  int g = blockIdx.x * blockDim.x + threadIdx.x;
  if (g >= NN * H) return;
  int n = g >> 7, c = g & 127;
  A[g] = (x[g] * scsh[c] + scsh[128 + c]) * dout_r[n];
}
// edge scatter-add: agg[dst] += A[src]; 32 threads/edge * 4 cols each
__global__ void k_scatter(const float* __restrict__ A, const int* __restrict__ src,
                          const int* __restrict__ dst, float* agg) {
  unsigned g = blockIdx.x * blockDim.x + threadIdx.x;
  int e = g >> 5;
  int c0 = (g & 31) << 2;
  int s = src[e], d = dst[e];
  const float4 v = *reinterpret_cast<const float4*>(A + (size_t)s * H + c0);
  float* o = agg + (size_t)d * H + c0;
  atomicAdd(o + 0, v.x); atomicAdd(o + 1, v.y);
  atomicAdd(o + 2, v.z); atomicAdd(o + 3, v.w);
}

// ---------- WMMA GEMM: out[N,NOUT] (+epilogue) = X[N,K] @ W[K,NOUT] ----------
// flags: 1 = relu(acc+bias); 2 = accumulate into out (out += ...)
// alpha: per-row scale applied to acc (after relu); resid added last.
// LDS holds W pre-swizzled so each lane's 16 B-elements are contiguous 32B:
//   sW[ ((k0/32)*NT + t)*512 + lane*16 + i ]  with  lane = kh*16 + (c%16),
//   element i of lane -> W[k0 + 16*kh + i][t*16 + c%16].
template<int K, int NOUT>
__global__ void k_gemm(const float* __restrict__ X, const float* __restrict__ rowscale,
                       const unsigned short* __restrict__ Wg, const float* __restrict__ bias,
                       const float* resid, const float* __restrict__ alpha,
                       float* out, int flags) {
  constexpr int NT = NOUT / 16;
  __shared__ unsigned short sW[K * NOUT];
  int tid = threadIdx.x;
  // cooperative swizzled stage: src index i = k*NOUT + c
  for (int i = tid; i < K * NOUT; i += 256) {
    int k = i / NOUT, c = i % NOUT;
    int kstep = k >> 5, krem = k & 31;
    int kh = krem >> 4, ki = krem & 15;
    int t = c >> 4, l = c & 15;
    sW[((kstep * NT + t) << 9) + ((kh << 4) + l) * 16 + ki] = Wg[i];
  }
  __syncthreads();

  int wave = tid >> 5, lane = tid & 31;
  int l16 = lane & 15, kh = lane >> 4;
  int row0 = blockIdx.x * 128 + wave * 16;

  v8f acc[NT];
  const v8f vz = {0.f, 0.f, 0.f, 0.f, 0.f, 0.f, 0.f, 0.f};
#pragma unroll
  for (int t = 0; t < NT; ++t) acc[t] = vz;

  const float* xr = X + (size_t)(row0 + l16) * K;
  float rs = rowscale ? rowscale[row0 + l16] : 1.0f;

#pragma unroll
  for (int k0 = 0; k0 < K; k0 += 32) {
    // A operand: 16x32 bf16 tile (contiguous k order, same as B in LDS):
    // lane(&15)=row; elems 0..7 -> K=k0+16*kh+j ; elems 8..15 -> K=k0+16*kh+8+j
    v16bf a;
    int ka = k0 + 16 * kh, kb = k0 + 16 * kh + 8;
#pragma unroll
    for (int j = 0; j < 8; ++j) {
      a[j]     = f2bf(xr[ka + j] * rs);
      a[8 + j] = f2bf(xr[kb + j] * rs);
    }
    const unsigned short* bbase = sW + ((k0 >> 5) * NT << 9) + lane * 16;
#pragma unroll
    for (int t = 0; t < NT; ++t) {
      const v8bf lo = *reinterpret_cast<const v8bf*>(bbase + (t << 9));
      const v8bf hi = *reinterpret_cast<const v8bf*>(bbase + (t << 9) + 8);
      v16bf bm = __builtin_shufflevector(lo, hi, 0, 1, 2, 3, 4, 5, 6, 7,
                                         8, 9, 10, 11, 12, 13, 14, 15);
      acc[t] = __builtin_amdgcn_wmma_f32_16x16x32_bf16(
          false, a, false, bm, (short)0, acc[t], false, false);
    }
  }

  float al[8];
  if (alpha) {
#pragma unroll
    for (int v = 0; v < 8; ++v) al[v] = alpha[row0 + 8 * kh + v];
  }
#pragma unroll
  for (int t = 0; t < NT; ++t) {
    int c = t * 16 + l16;
    float bs = bias ? bias[c] : 0.0f;
#pragma unroll
    for (int v = 0; v < 8; ++v) {
      float val = acc[t][v] + bs;
      if (flags & 1) val = fmaxf(val, 0.0f);
      if (alpha) val *= al[v];
      size_t o = (size_t)(row0 + 8 * kh + v) * NOUT + c;
      float base = 0.0f;
      if (resid) base = resid[o];
      if (flags & 2) base += out[o];
      out[o] = base + val;
    }
  }
}

// ---------- attention precompute (tiny) ----------
// kg[h,g,o] = sum_d inter_f[g,d] * Wk[h,d,o]
__global__ void k_kg(const float* __restrict__ inter_f, const float* __restrict__ Wk,
                     float* __restrict__ kg) {
  int g = blockIdx.x * blockDim.x + threadIdx.x;      // NHEAD*NB*H = 32768
  int hh = g >> 13, rem = g & 8191, gg = rem >> 7, o = rem & 127;
  float s = 0.f;
  for (int d = 0; d < H; ++d)
    s += inter_f[gg * H + d] * Wk[((hh * H + d) * H) + o];
  kg[g] = s;
}
// wvec[h,g,d] = sum_o Wq[h,d,o] * kg[h,g,o]
__global__ void k_wvec(const float* __restrict__ Wq, const float* __restrict__ kg,
                       float* __restrict__ wvec) {
  int g = blockIdx.x * blockDim.x + threadIdx.x;
  int hh = g >> 13, rem = g & 8191, gg = rem >> 7, d = rem & 127;
  float s = 0.f;
  const float* qr = Wq + (size_t)(hh * H + d) * H;
  const float* kr = kg + (size_t)(hh * NB + gg) * H;
  for (int o = 0; o < H; ++o) s += qr[o] * kr[o];
  wvec[g] = s;
}
// att[h,n] = (h2[n] . wvec[h, gid[n]]) / sqrt(1280)
__global__ void k_att(const float* __restrict__ h2, const float* __restrict__ wvec,
                      const int* __restrict__ gid, float* __restrict__ att) {
  int g = blockIdx.x * blockDim.x + threadIdx.x;      // NHEAD*NN
  int hh = g >> 16, n = g & 65535;
  const float* hr = h2 + (size_t)n * H;
  const float* wr = wvec + (size_t)(hh * NB + gid[n]) * H;
  float s = 0.f;
  for (int c = 0; c < H; ++c) s += hr[c] * wr[c];
  att[g] = s * 0.027950849718747373f;   // 1/sqrt(1280)
}
__global__ void k_amax(const float* __restrict__ att, const int* __restrict__ gid, unsigned* mx) {
  int g = blockIdx.x * blockDim.x + threadIdx.x;
  int hh = g >> 16, n = g & 65535;
  atomicMax(mx + hh * NB + gid[n], fenc(att[g]));
}
__global__ void k_aexp(float* att, const int* __restrict__ gid,
                       const unsigned* __restrict__ mx, float* ssum) {
  int g = blockIdx.x * blockDim.x + threadIdx.x;
  int hh = g >> 16, n = g & 65535;
  int i = hh * NB + gid[n];
  float e = __expf(att[g] - fdec(mx[i]));
  att[g] = e;
  atomicAdd(ssum + i, e);
}
__global__ void k_anorm(float* att, const int* __restrict__ gid, const float* __restrict__ ssum) {
  int g = blockIdx.x * blockDim.x + threadIdx.x;
  int hh = g >> 16, n = g & 65535;
  att[g] /= ssum[hh * NB + gid[n]];
}
// M_h = Wv[h] @ Wc_h  -> bf16
__global__ void k_mh(const float* __restrict__ Wv, const float* __restrict__ Wc,
                     unsigned short* __restrict__ Mhb) {
  int g = blockIdx.x * blockDim.x + threadIdx.x;      // NHEAD*H*H = 65536
  int hh = g >> 14, rem = g & 16383, d = rem >> 7, c = rem & 127;
  float s = 0.f;
  for (int o = 0; o < H; ++o)
    s += Wv[((hh * H + d) * H) + o] * Wc[((hh * H + o) * H) + c];
  Mhb[g] = bf2u(f2bf(s));
}

// ---------- LayerNorm (in place) and LN+segment-sum readout ----------
__global__ void k_ln(float* x, const float* __restrict__ g_, const float* __restrict__ b_) {
  int row = blockIdx.x, t = threadIdx.x;
  __shared__ float red[128];
  __shared__ float smean, srstd;
  float v = x[(size_t)row * H + t];
  red[t] = v; __syncthreads();
  for (int s = 64; s > 0; s >>= 1) { if (t < s) red[t] += red[t + s]; __syncthreads(); }
  if (t == 0) smean = red[0] * (1.0f / H);
  __syncthreads();
  float d = v - smean;
  red[t] = d * d; __syncthreads();
  for (int s = 64; s > 0; s >>= 1) { if (t < s) red[t] += red[t + s]; __syncthreads(); }
  if (t == 0) srstd = rsqrtf(red[0] * (1.0f / H) + EPS);
  __syncthreads();
  x[(size_t)row * H + t] = d * srstd * g_[t] + b_[t];
}
__global__ void k_lnread(const float* __restrict__ x, const float* __restrict__ g_,
                         const float* __restrict__ b_, const int* __restrict__ gid,
                         float* outp) {
  int row = blockIdx.x, t = threadIdx.x;
  __shared__ float red[128];
  __shared__ float smean, srstd;
  float v = x[(size_t)row * H + t];
  red[t] = v; __syncthreads();
  for (int s = 64; s > 0; s >>= 1) { if (t < s) red[t] += red[t + s]; __syncthreads(); }
  if (t == 0) smean = red[0] * (1.0f / H);
  __syncthreads();
  float d = v - smean;
  red[t] = d * d; __syncthreads();
  for (int s = 64; s > 0; s >>= 1) { if (t < s) red[t] += red[t + s]; __syncthreads(); }
  if (t == 0) srstd = rsqrtf(red[0] * (1.0f / H) + EPS);
  __syncthreads();
  float val = d * srstd * g_[t] + b_[t];
  atomicAdd(outp + gid[row] * H + t, val);
}

// ---------- driver ----------
extern "C" void kernel_launch(void* const* d_in, const int* in_sizes, int n_in,
                              void* d_out, int out_size, void* d_ws, size_t ws_size,
                              hipStream_t stream) {
  (void)in_sizes; (void)n_in; (void)out_size; (void)ws_size;
  const float* h      = (const float*)d_in[0];
  const float* interf = (const float*)d_in[1];
  const float* gamma1 = (const float*)d_in[2];
  const float* beta1  = (const float*)d_in[3];
  const float* W1     = (const float*)d_in[4];
  const float* b1     = (const float*)d_in[5];
  const float* gamma2 = (const float*)d_in[6];
  const float* beta2  = (const float*)d_in[7];
  const float* W2     = (const float*)d_in[8];
  const float* b2     = (const float*)d_in[9];
  const float* Wq     = (const float*)d_in[10];
  const float* Wk     = (const float*)d_in[11];
  const float* Wv     = (const float*)d_in[12];
  const float* Wc     = (const float*)d_in[13];
  const float* ln_g   = (const float*)d_in[14];
  const float* ln_b   = (const float*)d_in[15];
  const float* Wf1    = (const float*)d_in[16];
  const float* bf1    = (const float*)d_in[17];
  const float* Wf2    = (const float*)d_in[18];
  const float* bf2    = (const float*)d_in[19];
  const int*   src    = (const int*)d_in[20];
  const int*   dst    = (const int*)d_in[21];
  const int*   gid    = (const int*)d_in[22];
  float* out = (float*)d_out;                 // [readout 64*128][init_avg 64*128]

  const size_t NH = (size_t)NN * H;           // 8,388,608 floats (32 MB)
  float* wsf = (float*)d_ws;
  float* A   = wsf;                           // bn-normalized / scratch (overlaid by MID)
  float* Bb  = wsf + NH;                      // agg (overlaid by MID)
  float* C   = wsf + 2 * NH;                  // h1 -> h2 -> ffn2-out
  float* D   = wsf + 3 * NH;                  // mo -> x1
  float* MID = wsf;                           // [N, 2H] ffn hidden, overlays A+Bb
  float* smr = wsf + 4 * NH;                  // small region
  float* dout_r = smr;                        // NN
  float* din_r  = smr + NN;                   // NN
  float* stats  = din_r + NN;                 // 256
  float* scsh   = stats + 256;                // 256
  float* cnt    = scsh + 256;                 // 64
  unsigned* mx  = (unsigned*)(cnt + 64);      // 256
  float* ssum   = (float*)(mx + 256);         // 256
  float* kg     = ssum + 256;                 // 32768
  float* wvec   = kg + 32768;                 // 32768
  float* att    = wvec + 32768;               // 4*NN
  unsigned short* W1b  = (unsigned short*)(att + 4 * NN);
  unsigned short* W2b  = W1b + 16384;
  unsigned short* Wf1b = W2b + 16384;
  unsigned short* Wf2b = Wf1b + 32768;
  unsigned short* Mhb  = Wf2b + 32768;        // 4*16384

  // init outputs + small scratch
  k_zero<<<CDIV(2 * NB * H, 256), 256, 0, stream>>>(out, 2 * NB * H);
  k_zero<<<CDIV(2 * NN, 256), 256, 0, stream>>>(dout_r, 2 * NN);
  k_zero<<<1, 256, 0, stream>>>(stats, 256);
  k_zero<<<1, 64, 0, stream>>>(cnt, 64);
  k_zero<<<2, 256, 0, stream>>>((float*)mx, 512);   // mx + ssum

  // init pooling (mean_nodes)
  k_count<<<CDIV(NN, 256), 256, 0, stream>>>(gid, cnt);
  k_pool<<<CDIV(NN * H, 256), 256, 0, stream>>>(h, gid, out + NB * H);
  k_pooldiv<<<CDIV(NB * H, 256), 256, 0, stream>>>(out + NB * H, cnt);

  // degrees -> rsqrt(max(deg,1))
  k_degree<<<CDIV(NE, 256), 256, 0, stream>>>(src, dst, dout_r, din_r);
  k_rsqrt<<<CDIV(2 * NN, 256), 256, 0, stream>>>(dout_r, 2 * NN);

  // weight conversions
  k_cvt_bf16<<<CDIV(H * H, 256), 256, 0, stream>>>(W1, W1b, H * H);
  k_cvt_bf16<<<CDIV(H * H, 256), 256, 0, stream>>>(W2, W2b, H * H);
  k_cvt_bf16<<<CDIV(H * 2 * H, 256), 256, 0, stream>>>(Wf1, Wf1b, H * 2 * H);
  k_cvt_bf16<<<CDIV(2 * H * H, 256), 256, 0, stream>>>(Wf2, Wf2b, 2 * H * H);

  // ---- conv block 1: h1 = h + relu(gconv(bn(h)) @ W1 + b1) ----
  k_bnstats<<<CDIV((NN / 64) * H, 256), 256, 0, stream>>>(h, stats);
  k_bnfinal<<<1, 128, 0, stream>>>(stats, gamma1, beta1, scsh);
  k_bnapply<<<CDIV(NN * H, 256), 256, 0, stream>>>(h, scsh, dout_r, A);
  k_zero<<<CDIV((int)NH, 256), 256, 0, stream>>>(Bb, (int)NH);
  k_scatter<<<CDIV(NE * 32, 256), 256, 0, stream>>>(A, src, dst, Bb);
  k_gemm<128, 128><<<NN / 128, 256, 0, stream>>>(Bb, din_r, W1b, b1, h, nullptr, C, 1);

  // ---- conv block 2: h2 = h1 + relu(gconv(bn(h1)) @ W2 + b2) (in-place on C) ----
  k_zero<<<1, 256, 0, stream>>>(stats, 256);
  k_bnstats<<<CDIV((NN / 64) * H, 256), 256, 0, stream>>>(C, stats);
  k_bnfinal<<<1, 128, 0, stream>>>(stats, gamma2, beta2, scsh);
  k_bnapply<<<CDIV(NN * H, 256), 256, 0, stream>>>(C, scsh, dout_r, A);
  k_zero<<<CDIV((int)NH, 256), 256, 0, stream>>>(Bb, (int)NH);
  k_scatter<<<CDIV(NE * 32, 256), 256, 0, stream>>>(A, src, dst, Bb);
  k_gemm<128, 128><<<NN / 128, 256, 0, stream>>>(Bb, din_r, W2b, b2, C, nullptr, C, 1);

  // ---- attention: att scores + segment softmax ----
  k_kg<<<CDIV(NHEAD * NB * H, 256), 256, 0, stream>>>(interf, Wk, kg);
  k_wvec<<<CDIV(NHEAD * NB * H, 256), 256, 0, stream>>>(Wq, kg, wvec);
  k_att<<<CDIV(NHEAD * NN, 256), 256, 0, stream>>>(C, wvec, gid, att);
  k_amax<<<CDIV(NHEAD * NN, 256), 256, 0, stream>>>(att, gid, mx);
  k_aexp<<<CDIV(NHEAD * NN, 256), 256, 0, stream>>>(att, gid, mx, ssum);
  k_anorm<<<CDIV(NHEAD * NN, 256), 256, 0, stream>>>(att, gid, ssum);
  k_mh<<<CDIV(NHEAD * H * H, 256), 256, 0, stream>>>(Wv, Wc, Mhb);

  // ---- mo + h2 : D = h2 + sum_h alpha_h * (h2 @ M_h) ----
  k_gemm<128, 128><<<NN / 128, 256, 0, stream>>>(C, nullptr, Mhb, nullptr, C, att, D, 0);
  for (int hh = 1; hh < NHEAD; ++hh)
    k_gemm<128, 128><<<NN / 128, 256, 0, stream>>>(C, nullptr, Mhb + hh * H * H, nullptr,
                                                   nullptr, att + (size_t)hh * NN, D, 2);
  // x1 = LN(mo + h2)  (in place on D)
  k_ln<<<NN, 128, 0, stream>>>(D, ln_g, ln_b);

  // ---- FFN: MID = relu(x1@Wf1+bf1); C = MID@Wf2 + bf2 + x1 ----
  k_gemm<128, 256><<<NN / 128, 256, 0, stream>>>(D, nullptr, Wf1b, bf1, nullptr, nullptr, MID, 1);
  k_gemm<256, 128><<<NN / 128, 256, 0, stream>>>(MID, nullptr, Wf2b, bf2, D, nullptr, C, 0);

  // x2 = LN(C); readout = segment_sum(x2)
  k_lnread<<<NN, 128, 0, stream>>>(C, ln_g, ln_b, gid, out);
}
